// BlockModel_9758165696627
// MI455X (gfx1250) — matrix operand order repaired and checked by reading, not compiled, not run on hardware
//
#include <hip/hip_runtime.h>

#define D_FEAT 129
#define H_DIM  128
#define IN_DIM 259
#define KPAD1  288      // 259 padded to 9 K-tiles of 32
#define NPAD_O 144      // 129 padded to 9 N-tiles of 16

// per-block staged weight blob: [W1t 288x128 | W2t 128x128] col-major f16
#define W1_HALVES (KPAD1 * H_DIM)            // 36864
#define W2_HALVES (H_DIM * H_DIM)            // 16384
#define WBLK_HALVES (W1_HALVES + W2_HALVES)  // 53248
#define WBLK_BYTES  (WBLK_HALVES * 2)        // 106496
#define WBLK_QW     (WBLK_BYTES / 8)         // 13312  (fits 16-bit tile_dim0)

typedef __attribute__((ext_vector_type(16))) _Float16 v16h;
typedef __attribute__((ext_vector_type(8)))  _Float16 v8h;
typedef __attribute__((ext_vector_type(8)))  float    v8f;
typedef unsigned int u32x4 __attribute__((ext_vector_type(4)));
typedef int          i32x4 __attribute__((ext_vector_type(4)));
typedef int          i32x8 __attribute__((ext_vector_type(8)));

#if defined(__gfx1250__) && __has_builtin(__builtin_amdgcn_tensor_load_to_lds)
#define USE_TDM 1
#endif

// ---- WMMA fragment loaders (wave32 layouts, ISA 7.12.2) -------------------
// A (16x32 f16): lane L holds row M=L&15; halves 0..7 = K kbase+8*(L>>4)..,
// halves 8..15 = K kbase+16+8*(L>>4)..  -> two 16B chunks per lane.
__device__ __forceinline__ v16h frag_a(const _Float16* row, int kbase, int hi) {
  const _Float16* p = row + kbase + hi * 8;
  v8h lo = *(const v8h*)(p);
  v8h hh = *(const v8h*)(p + 16);
  return __builtin_shufflevector(lo, hh, 0,1,2,3,4,5,6,7,8,9,10,11,12,13,14,15);
}
// B (32x16 f16): lane L holds column N=L&15; halves 0..7 = K kbase+8*(L>>4)..,
// halves 8..15 = K kbase+16+8*(L>>4)..  (same order as A).
// Weights stored column-major (K contiguous per column), 32B aligned.
__device__ __forceinline__ v16h frag_b(const _Float16* w, int col, int kbase,
                                       int hi, int kpad) {
  const _Float16* p = w + (size_t)col * kpad + kbase + hi * 8;
  v8h lo = *(const v8h*)(p);
  v8h hh = *(const v8h*)(p + 16);
  return __builtin_shufflevector(lo, hh, 0,1,2,3,4,5,6,7,8,9,10,11,12,13,14,15);
}

__device__ __forceinline__ v8f wmma_f16(v16h a, v16h b, v8f c) {
  return __builtin_amdgcn_wmma_f32_16x16x32_f16(false, a, false, b,
                                                (short)0, c, false, false);
}

// ---- TDM staging: one wave DMAs the whole weight blob into LDS ------------
// D# per ISA 8.3/8.4: 1-D tile, data_size=8B, count=1, type=2("image").
__device__ __forceinline__ void tdm_stage_weights(const _Float16* gsrc,
                                                  _Float16* lds_dst) {
#ifdef USE_TDM
  unsigned lds_off = (unsigned)(size_t)(void*)lds_dst;   // addr[31:0] = LDS offset
  unsigned long long ga = (unsigned long long)(size_t)gsrc;
  u32x4 g0;
  g0[0] = 1u;                                   // count=1, load, user mode
  g0[1] = lds_off;                              // lds_addr
  g0[2] = (unsigned)(ga & 0xFFFFFFFFu);         // global_addr[31:0]
  g0[3] = (unsigned)((ga >> 32) & 0x01FFFFFFu)  // global_addr[56:32]
          | (2u << 30);                         // type=2
  i32x8 g1 = {0, 0, 0, 0, 0, 0, 0, 0};
  g1[0] = 0x30000;                              // data_size=3 (8 bytes)
  g1[1] = (int)((WBLK_QW & 0xFFFF) << 16);      // tensor_dim0[15:0]
  g1[2] = (int)(WBLK_QW >> 16);                 // tensor_dim0[31:16]
  g1[3] = (int)((WBLK_QW & 0xFFFF) << 16);      // tile_dim0
  g1[5] = (int)WBLK_QW;                         // tensor_dim0_stride[31:0]
  i32x4 gz = {0, 0, 0, 0};
#if __clang_major__ >= 23
  i32x8 gz8 = {0, 0, 0, 0, 0, 0, 0, 0};
  __builtin_amdgcn_tensor_load_to_lds(g0, g1, gz, gz, gz8, 0);
#else
  __builtin_amdgcn_tensor_load_to_lds(g0, g1, gz, gz, 0);
#endif
#else
  (void)gsrc; (void)lds_dst;
#endif
}

// ---- weight prep: fp32 -> f16, transpose to col-major, zero-pad K/N -------
__global__ void prep_weights(const float* __restrict__ b0W1, const float* __restrict__ b0W2,
                             const float* __restrict__ b0Wo, const float* __restrict__ b1W1,
                             const float* __restrict__ b1W2, const float* __restrict__ b1Wo,
                             _Float16* __restrict__ Wb, _Float16* __restrict__ Wot) {
  int idx = blockIdx.x * blockDim.x + threadIdx.x;
  const int n1 = 2 * W1_HALVES;       // 73728
  const int n2 = 2 * W2_HALVES;       // 32768
  const int n3 = 2 * NPAD_O * H_DIM;  // 36864
  if (idx < n1) {
    int b = idx / W1_HALVES, r = idx % W1_HALVES;
    int col = r / KPAD1, k = r % KPAD1;
    const float* W = b ? b1W1 : b0W1;
    float v = (k < IN_DIM) ? W[(size_t)k * H_DIM + col] : 0.f;
    Wb[(size_t)b * WBLK_HALVES + (size_t)col * KPAD1 + k] = (_Float16)v;
  } else if ((idx -= n1) < n2) {
    int b = idx / W2_HALVES, r = idx % W2_HALVES;
    int col = r / H_DIM, k = r % H_DIM;
    const float* W = b ? b1W2 : b0W2;
    Wb[(size_t)b * WBLK_HALVES + W1_HALVES + (size_t)col * H_DIM + k] =
        (_Float16)W[(size_t)k * H_DIM + col];
  } else if ((idx -= n2) < n3) {
    int b = idx / (NPAD_O * H_DIM), r = idx % (NPAD_O * H_DIM);
    int col = r / H_DIM, k = r % H_DIM;
    const float* W = b ? b1Wo : b0Wo;
    float v = (col < D_FEAT) ? W[(size_t)k * D_FEAT + col] : 0.f;
    Wot[(size_t)b * NPAD_O * H_DIM + (size_t)col * H_DIM + k] = (_Float16)v;
  }
}

__global__ void zero_i32(int* __restrict__ p, int n) {
  int i = blockIdx.x * blockDim.x + threadIdx.x;
  if (i < n) p[i] = 0;
}

// ---- edge MLP: gather -> relu(x@W1) -> relu(h@W2) -> fused segment-max ----
// 4 waves/WG, 1 wave = 16 edges x 128 outputs; weights LDS-resident (TDM).
// Dynamic LDS: W1s 72KB | W2s 32KB | sx 36KB | sh 16KB = 156KB (2 WG / WGP).
__global__ __launch_bounds__(128) void edge_mlp(
    const float* __restrict__ interp, const float* __restrict__ add_info,
    const int* __restrict__ nbr_idx, const int* __restrict__ self_idx,
    const _Float16* __restrict__ Wblk, const float* __restrict__ bias1,
    const float* __restrict__ bias2, int* __restrict__ pooled, int E) {
  extern __shared__ int4 smem4[];
  _Float16* smem = (_Float16*)smem4;
  _Float16* W1s = smem;                       // 36864 halves
  _Float16* W2s = smem + W1_HALVES;           // 16384 halves
  _Float16* sx  = smem + WBLK_HALVES;         // 4*16*288 halves
  _Float16* sh  = sx + 4 * 16 * KPAD1;        // 4*16*128 halves

  const int lane = threadIdx.x & 31;
  const int wv   = threadIdx.x >> 5;
  const int m16  = lane & 15;
  const int hi   = lane >> 4;
  const int e0   = (blockIdx.x * 4 + wv) * 16;

  // ---- stage weights into LDS (TDM on wave 0; async vs gather below) -----
#ifdef USE_TDM
  if (threadIdx.x < 32) tdm_stage_weights(Wblk, W1s);
#else
  for (int i = threadIdx.x; i < WBLK_BYTES / 16; i += 128)
    ((int4*)W1s)[i] = ((const int4*)Wblk)[i];
#endif

  // ---- build x tile: [nbr(129) | main(129) | iou(1) | pad(29)] as f16 ----
  _Float16* sxw = sx + (size_t)wv * 16 * KPAD1;
  for (int r = 0; r < 16; ++r) {
    int e = e0 + r; e = (e < E) ? e : (E - 1);
    const float* nrow = interp + (size_t)nbr_idx[e] * D_FEAT;
    const float* srow = interp + (size_t)self_idx[e] * D_FEAT;
    float iou = add_info[e];
    #pragma unroll
    for (int j = 0; j < 9; ++j) {
      int c = lane + j * 32;
      float v;
      if (c < D_FEAT)            v = nrow[c];
      else if (c < 2 * D_FEAT)   v = srow[c - D_FEAT];
      else if (c == 2 * D_FEAT)  v = iou;
      else                       v = 0.f;
      sxw[r * KPAD1 + c] = (_Float16)v;
    }
  }
#ifdef USE_TDM
  if (threadIdx.x < 32) __builtin_amdgcn_s_wait_tensorcnt(0);
#endif
  __syncthreads();

  const v8f vzero = {0.f, 0.f, 0.f, 0.f, 0.f, 0.f, 0.f, 0.f};
  v8f acc[8];
  #pragma unroll
  for (int n = 0; n < 8; ++n) acc[n] = vzero;

  // ---- layer 1: (16x288) @ (288x128), B from LDS, batched loads ----------
  const _Float16* arow = sxw + m16 * KPAD1;
  #pragma unroll
  for (int k = 0; k < 9; ++k) {
    v16h a = frag_a(arow, k * 32, hi);
    v16h b[8];
    #pragma unroll
    for (int n = 0; n < 8; ++n) b[n] = frag_b(W1s, n * 16 + m16, k * 32, hi, KPAD1);
    #pragma unroll
    for (int n = 0; n < 8; ++n) acc[n] = wmma_f16(a, b[n], acc[n]);
  }
  // bias + relu -> f16 h1 tile in LDS (C layout: lane=col tile, row=r+8*hi)
  _Float16* shw = sh + (size_t)wv * 16 * H_DIM;
  #pragma unroll
  for (int n = 0; n < 8; ++n) {
    float bb = bias1[n * 16 + m16];
    #pragma unroll
    for (int r = 0; r < 8; ++r) {
      float v = fmaxf(acc[n][r] + bb, 0.f);
      shw[(r + 8 * hi) * H_DIM + n * 16 + m16] = (_Float16)v;
    }
  }
  __syncthreads();

  // ---- layer 2: (16x128) @ (128x128), B from LDS -------------------------
  #pragma unroll
  for (int n = 0; n < 8; ++n) acc[n] = vzero;
  const _Float16* arow2 = shw + m16 * H_DIM;
  #pragma unroll
  for (int k = 0; k < 4; ++k) {
    v16h a = frag_a(arow2, k * 32, hi);
    v16h b[8];
    #pragma unroll
    for (int n = 0; n < 8; ++n) b[n] = frag_b(W2s, n * 16 + m16, k * 32, hi, H_DIM);
    #pragma unroll
    for (int n = 0; n < 8; ++n) acc[n] = wmma_f16(a, b[n], acc[n]);
  }

  // ---- bias + relu + fused ragged segment-max via int-bits atomicMax -----
  // relu output >= 0 and pooled zero-initialized => int compare on the bit
  // pattern == float max, and every segment is non-empty.
  int  seg[8];
  bool ok[8];
  #pragma unroll
  for (int r = 0; r < 8; ++r) {
    int e = e0 + r + 8 * hi;
    ok[r]  = (e < E);
    seg[r] = self_idx[ok[r] ? e : 0];
  }
  #pragma unroll
  for (int n = 0; n < 8; ++n) {
    float bb = bias2[n * 16 + m16];
    #pragma unroll
    for (int r = 0; r < 8; ++r) {
      if (ok[r]) {
        float v = fmaxf(acc[n][r] + bb, 0.f);
        atomicMax(pooled + (size_t)seg[r] * H_DIM + n * 16 + m16,
                  __float_as_int(v));
      }
    }
  }
}

// ---- pooled @ Wo + bo + residual (elementwise, in-place safe) -------------
__global__ __launch_bounds__(128) void pooled_gemm_residual(
    const float* __restrict__ pooledf, const _Float16* __restrict__ Wot,
    const float* __restrict__ bo, const float* __restrict__ src,
    float* __restrict__ dst, int nRows) {
  __shared__ __align__(16) _Float16 sp[4][16][H_DIM];       // 16 KB
  __shared__ __align__(16) _Float16 wos[NPAD_O * H_DIM];    // 36 KB
  const int lane = threadIdx.x & 31;
  const int wv   = threadIdx.x >> 5;
  const int m16  = lane & 15;
  const int hi   = lane >> 4;
  const int row0 = (blockIdx.x * 4 + wv) * 16;

  for (int i = threadIdx.x; i < (NPAD_O * H_DIM) / 8; i += 128)
    ((int4*)wos)[i] = ((const int4*)Wot)[i];

  for (int r = 0; r < 16; ++r) {
    int rr = row0 + r; rr = (rr < nRows) ? rr : (nRows - 1);
    #pragma unroll
    for (int j = 0; j < 4; ++j) {
      int c = lane + j * 32;
      sp[wv][r][c] = (_Float16)pooledf[(size_t)rr * H_DIM + c];
    }
  }
  __syncthreads();

  const v8f vzero = {0.f, 0.f, 0.f, 0.f, 0.f, 0.f, 0.f, 0.f};
  v8f acc[9];
  #pragma unroll
  for (int n = 0; n < 9; ++n) acc[n] = vzero;

  const _Float16* arow = &sp[wv][m16][0];
  #pragma unroll
  for (int k = 0; k < 4; ++k) {
    v16h a = frag_a(arow, k * 32, hi);
    v16h b[9];
    #pragma unroll
    for (int n = 0; n < 9; ++n) b[n] = frag_b(wos, n * 16 + m16, k * 32, hi, H_DIM);
    #pragma unroll
    for (int n = 0; n < 9; ++n) acc[n] = wmma_f16(a, b[n], acc[n]);
  }
  #pragma unroll
  for (int n = 0; n < 9; ++n) {
    int col = n * 16 + m16;
    if (col < D_FEAT) {
      float bb = bo[col];
      #pragma unroll
      for (int r = 0; r < 8; ++r) {
        int row = row0 + r + 8 * hi;
        if (row < nRows) {
          dst[(size_t)row * D_FEAT + col] =
              src[(size_t)row * D_FEAT + col] + acc[n][r] + bb;
        }
      }
    }
  }
}

// ---- final head: relu(x@fW1+fb1)@fW2+fb2, one wave per row ----------------
__global__ __launch_bounds__(128) void final_head(
    const float* __restrict__ src, const float* __restrict__ fW1,
    const float* __restrict__ fb1, const float* __restrict__ fW2,
    const float* __restrict__ fb2, float* __restrict__ out, int nRows) {
  const int lane = threadIdx.x & 31;
  const int row  = blockIdx.x * 4 + (threadIdx.x >> 5);
  if (row >= nRows) return;
  const float* x = src + (size_t)row * D_FEAT;
  float h0 = fb1[lane], h1 = fb1[lane + 32];
  for (int c = 0; c < D_FEAT; ++c) {
    float xv = x[c];
    h0 = fmaf(xv, fW1[c * 64 + lane], h0);
    h1 = fmaf(xv, fW1[c * 64 + lane + 32], h1);
  }
  float part = fmaxf(h0, 0.f) * fW2[lane] + fmaxf(h1, 0.f) * fW2[lane + 32];
  #pragma unroll
  for (int off = 16; off > 0; off >>= 1) part += __shfl_down(part, off, 32);
  if (lane == 0) out[row] = part + fb2[0];
}

extern "C" void kernel_launch(void* const* d_in, const int* in_sizes, int n_in,
                              void* d_out, int out_size, void* d_ws, size_t ws_size,
                              hipStream_t stream) {
  const float* interp   = (const float*)d_in[0];
  const float* add_info = (const float*)d_in[1];
  // d_in[2] = neighborhood_sizes (unused: segment-max fused via atomics)
  const int* nbr  = (const int*)d_in[3];
  const int* self = (const int*)d_in[4];
  const float* b0W1 = (const float*)d_in[5],  *b0b1 = (const float*)d_in[6];
  const float* b0W2 = (const float*)d_in[7],  *b0b2 = (const float*)d_in[8];
  const float* b0Wo = (const float*)d_in[9],  *b0bo = (const float*)d_in[10];
  const float* b1W1 = (const float*)d_in[11], *b1b1 = (const float*)d_in[12];
  const float* b1W2 = (const float*)d_in[13], *b1b2 = (const float*)d_in[14];
  const float* b1Wo = (const float*)d_in[15], *b1bo = (const float*)d_in[16];
  const float* fW1  = (const float*)d_in[17], *fb1  = (const float*)d_in[18];
  const float* fW2  = (const float*)d_in[19], *fb2  = (const float*)d_in[20];

  const int E = in_sizes[1];             // 253952
  const int N = in_sizes[0] / D_FEAT;    // 15872

  // workspace layout (64B-aligned offsets); total ~16.6 MB
  char* ws = (char*)d_ws;
  _Float16* Wb  = (_Float16*)(ws);            // 2 * 53248 f16  (W1|W2 per block)
  _Float16* Wot = (_Float16*)(ws + 212992);   // 2 * 18432 f16
  int*   pooled  = (int*)(ws + 286720);       // N*128 i32
  float* interpA = (float*)(ws + 286720 + (size_t)N * H_DIM * 4);

  prep_weights<<<(143360 + 255) / 256, 256, 0, stream>>>(
      b0W1, b0W2, b0Wo, b1W1, b1W2, b1Wo, Wb, Wot);

  const int pn    = N * H_DIM;
  const int egrid = ((E + 15) / 16 + 3) / 4;   // 4 waves/WG, 16 edges/wave
  const int pgrid = ((N + 15) / 16 + 3) / 4;
  const size_t edge_lds = (size_t)(WBLK_HALVES + 4 * 16 * KPAD1 + 4 * 16 * H_DIM) * 2;

  // ---- block 0 ----
  zero_i32<<<(pn + 255) / 256, 256, 0, stream>>>(pooled, pn);
  edge_mlp<<<egrid, 128, edge_lds, stream>>>(interp, add_info, nbr, self,
                                             Wb, b0b1, b0b2, pooled, E);
  pooled_gemm_residual<<<pgrid, 128, 0, stream>>>(
      (const float*)pooled, Wot, b0bo, interp, interpA, N);

  // ---- block 1 ----
  zero_i32<<<(pn + 255) / 256, 256, 0, stream>>>(pooled, pn);
  edge_mlp<<<egrid, 128, edge_lds, stream>>>(interpA, add_info, nbr, self,
                                             Wb + WBLK_HALVES, b1b1, b1b2,
                                             pooled, E);
  pooled_gemm_residual<<<pgrid, 128, 0, stream>>>(
      (const float*)pooled, Wot + NPAD_O * H_DIM, b1bo, interpA, interpA, N);

  // ---- final head ----
  final_head<<<(N + 3) / 4, 128, 0, stream>>>(interpA, fW1, fb1, fW2, fb2,
                                              (float*)d_out, N);
}
